// SpatialTemporalAttention_42039139893702
// MI455X (gfx1250) — matrix operand order repaired and checked by reading, not compiled, not run on hardware
//
#include <hip/hip_runtime.h>

typedef __attribute__((ext_vector_type(16))) __bf16 v16bf;
typedef __attribute__((ext_vector_type(8)))  __bf16 v8bf;
typedef __attribute__((ext_vector_type(8)))  float  v8f;

#define DEV static __device__ __forceinline__

constexpr int BATCH = 2;
constexpr int NTOK  = 4096;            // t*h*w
constexpr int CCH   = 256;             // channels
constexpr int NH    = 8;               // heads
constexpr int DH    = 64;              // dim head
constexpr int INNER = NH * DH;         // 512
constexpr int ROWS  = BATCH * NTOK;    // 8192

constexpr int KLP = 72;                // padded LDS row stride for K block (64+8)
constexpr int VLP = 40;                // padded LDS row stride for V block (32+8)

DEV v8f vzero() {
  v8f z;
#pragma unroll
  for (int i = 0; i < 8; ++i) z[i] = 0.f;
  return z;
}
DEV v16bf cat16(v8bf a, v8bf b) {
  return __builtin_shufflevector(a, b, 0,1,2,3,4,5,6,7,8,9,10,11,12,13,14,15);
}
DEV v16bf ld16(const __bf16* p) { return *reinterpret_cast<const v16bf*>(p); }
DEV v8bf  ld8 (const __bf16* p) { return *reinterpret_cast<const v8bf*>(p); }
DEV void  st8 (__bf16* p, v8bf v) { *reinterpret_cast<v8bf*>(p) = v; }
DEV v16bf ld16u(const __bf16* p) { return cat16(ld8(p), ld8(p + 8)); } // 16B-aligned pair
DEV v8f wmma_bf16(v16bf a, v16bf b, v8f c) {
  return __builtin_amdgcn_wmma_f32_16x16x32_bf16(false, a, false, b, (short)0, c, false, false);
}

// ---------------- stage 0: fp32 -> bf16 conversion ----------------
__global__ void k_cvt_bf16(const float* __restrict__ in, __bf16* __restrict__ out, int n) {
  int i = blockIdx.x * blockDim.x + threadIdx.x;
  if (i < n) out[i] = (__bf16)in[i];
}

// ---------------- stage 1: QKV projection (y = x * W^T) ----------------
// One wave -> one 16x64 output tile (4 n-tiles share each A operand).
// Software-pipelined k-loop: next-step A/B loads overlap current-step WMMAs.
// A 64-wide e-group is exactly one head. Q,K stored [b,h,N,dh]; V transposed [b,h,dh,N].
__global__ void __launch_bounds__(256) k_qkv(const __bf16* __restrict__ xb,
                                             const __bf16* __restrict__ Wqb,
                                             const __bf16* __restrict__ Wkb,
                                             const __bf16* __restrict__ Wvb,
                                             __bf16* __restrict__ Qb,
                                             __bf16* __restrict__ Kb,
                                             __bf16* __restrict__ Vtb) {
  const int lane = threadIdx.x & 31;
  const int wid  = blockIdx.x * 8 + (threadIdx.x >> 5);
  const int proj = wid / (512 * 8);           // 0=Q 1=K 2=V
  const int rem  = wid % (512 * 8);
  const int rowtile = rem >> 3;               // 512 row tiles over 8192 rows
  const int eg      = rem & 7;                // e-group of 64 == head index
  const int m = lane & 15, half = lane >> 4;

  const __bf16* Wsrc = (proj == 0) ? Wqb : ((proj == 1) ? Wkb : Wvb);
  const __bf16* xr = xb + (size_t)(rowtile * 16 + m) * CCH;
  const __bf16* wr = Wsrc + (size_t)(eg * 64 + m) * CCH + half * 16;

  v8f acc[4];
#pragma unroll
  for (int t = 0; t < 4; ++t) acc[t] = vzero();

  // pipeline prologue
  v16bf a = ld16(xr + half * 16);
  v16bf b[4];
#pragma unroll
  for (int t = 0; t < 4; ++t) b[t] = ld16(wr + (size_t)t * 16 * CCH);

#pragma unroll
  for (int cc = 0; cc < CCH; cc += 32) {
    v16bf an, bn[4];
    if (cc + 32 < CCH) {
      an = ld16(xr + cc + 32 + half * 16);
#pragma unroll
      for (int t = 0; t < 4; ++t) bn[t] = ld16(wr + (size_t)t * 16 * CCH + cc + 32);
    }
#pragma unroll
    for (int t = 0; t < 4; ++t) acc[t] = wmma_bf16(a, b[t], acc[t]);
    if (cc + 32 < CCH) {
      a = an;
#pragma unroll
      for (int t = 0; t < 4; ++t) b[t] = bn[t];
    }
  }

  const int row0 = rowtile * 16;
  const int bb = row0 >> 12, n0 = row0 & (NTOK - 1);
  const int head = eg;

  if (proj < 2) {
    __bf16* dst = ((proj == 0) ? Qb : Kb) + ((size_t)(bb * NH + head) * NTOK) * DH;
#pragma unroll
    for (int t = 0; t < 4; ++t)
#pragma unroll
      for (int r = 0; r < 8; ++r)
        dst[(size_t)(n0 + r + 8 * half) * DH + t * 16 + m] = (__bf16)acc[t][r];
  } else {
    // transposed store: [dh][N], 8 consecutive n per lane -> one 16B store per tile
#pragma unroll
    for (int t = 0; t < 4; ++t) {
      __bf16* dst = Vtb + ((size_t)(bb * NH + head) * DH + t * 16 + m) * NTOK + n0 + 8 * half;
      v8bf o;
#pragma unroll
      for (int r = 0; r < 8; ++r) o[r] = (__bf16)acc[t][r];
      st8(dst, o);
    }
  }
}

// ---------------- stage 2: flash attention ----------------
// Block = 8 waves on 8 consecutive 16-query tiles of one (b,head).
// K/V blocks of 32 keys staged through double-buffered LDS, shared by all waves.
__global__ void __launch_bounds__(256) k_attn(const __bf16* __restrict__ Qb,
                                              const __bf16* __restrict__ Kb,
                                              const __bf16* __restrict__ Vtb,
                                              __bf16* __restrict__ Ob) {
  __shared__ __align__(32) __bf16 Kl[2][32][KLP];   // 9.2 KB
  __shared__ __align__(32) __bf16 Vl[2][64][VLP];   // 10.2 KB
  __shared__ __align__(32) __bf16 Pl[8][16 * 32];   // 8 KB per-wave P staging

  const int tid  = threadIdx.x;
  const int lane = tid & 31;
  const int wib  = tid >> 5;
  const int bh     = blockIdx.x >> 5;               // b*8 + head
  const int qgroup = blockIdx.x & 31;
  const int qtile  = qgroup * 8 + wib;
  const int m = lane & 15, half = lane >> 4;
  const float scale = 0.125f;                       // 64^-0.5
  const float L2E = 1.4426950408889634f;

  const __bf16* Qh = Qb  + ((size_t)bh * NTOK) * DH;
  const __bf16* Kh = Kb  + ((size_t)bh * NTOK) * DH;
  const __bf16* Vh = Vtb + ((size_t)bh * DH) * NTOK;

  // Q tile as A operand (two K=32 chunks over dh=64)
  v16bf qa[2];
  {
    const __bf16* qr = Qh + (size_t)(qtile * 16 + m) * DH;
#pragma unroll
    for (int u = 0; u < 2; ++u)
      qa[u] = ld16(qr + u * 32 + half * 16);
  }

  // cooperative-load assignments: K block 32x64, V block 64x32 (both 4KB)
  const int krow = tid >> 3, kcol = (tid & 7) * 8;  // K: row 0..31, 8-elem chunk
  const int vrow = tid >> 2, vcol = (tid & 3) * 8;  // V: row 0..63, 8-elem chunk
  const __bf16* kg = Kh + (size_t)krow * DH + kcol;
  const __bf16* vg = Vh + (size_t)vrow * NTOK + vcol;

  float mrow[8], lrow[8];
  v8f acc[4];
#pragma unroll
  for (int r = 0; r < 8; ++r) { mrow[r] = -1e30f; lrow[r] = 0.f; }
#pragma unroll
  for (int t = 0; t < 4; ++t) acc[t] = vzero();

  __bf16* pw = Pl[wib];

  // prologue: stage key block 0
  {
    v8bf kr = ld8(kg), vr = ld8(vg);
    st8(&Kl[0][krow][kcol], kr);
    st8(&Vl[0][vrow][vcol], vr);
  }
  __syncthreads();

  for (int j = 0; j < NTOK / 32; ++j) {
    const int cur = j & 1, nxt = cur ^ 1;
    // prefetch next key block into registers (overlaps with WMMA below)
    v8bf kr, vr;
    const bool more = (j + 1) < NTOK / 32;
    if (more) {
      kr = ld8(kg + (size_t)(j + 1) * 32 * DH);
      vr = ld8(vg + (j + 1) * 32);
    }

    // S = Q K^T : two 16x16 f32 tiles (keys 0..15, 16..31 of block)
    v8f s0 = vzero(), s1 = vzero();
#pragma unroll
    for (int u = 0; u < 2; ++u) {
      v16bf b0 = ld16u(&Kl[cur][m][u * 32 + half * 16]);
      v16bf b1 = ld16u(&Kl[cur][16 + m][u * 32 + half * 16]);
      s0 = wmma_bf16(qa[u], b0, s0);
      s1 = wmma_bf16(qa[u], b1, s1);
    }

    // V B-operands: issue LDS loads early so they hide under softmax VALU
    v16bf bv[4];
#pragma unroll
    for (int t = 0; t < 4; ++t) bv[t] = ld16u(&Vl[cur][t * 16 + m][half * 16]);

    // --- online softmax (row r of this half-wave lives in VGPR r, cols in lanes) ---
    float bm[8];
#pragma unroll
    for (int r = 0; r < 8; ++r) {
      float a0 = s0[r] * scale, a1 = s1[r] * scale;
      s0[r] = a0; s1[r] = a1;
      bm[r] = fmaxf(a0, a1);
    }
    for (int off = 1; off < 16; off <<= 1)
#pragma unroll
      for (int r = 0; r < 8; ++r)
        bm[r] = fmaxf(bm[r], __shfl_xor(bm[r], off, 32));
    float alpha[8];
#pragma unroll
    for (int r = 0; r < 8; ++r) {
      float mn = fmaxf(mrow[r], bm[r]);
      alpha[r] = __builtin_amdgcn_exp2f((mrow[r] - mn) * L2E);
      mrow[r] = mn;
    }
    float rs[8];
#pragma unroll
    for (int r = 0; r < 8; ++r) {
      float p0 = __builtin_amdgcn_exp2f((s0[r] - mrow[r]) * L2E);
      float p1 = __builtin_amdgcn_exp2f((s1[r] - mrow[r]) * L2E);
      s0[r] = p0; s1[r] = p1;
      rs[r] = p0 + p1;
    }
    for (int off = 1; off < 16; off <<= 1)
#pragma unroll
      for (int r = 0; r < 8; ++r)
        rs[r] += __shfl_xor(rs[r], off, 32);
#pragma unroll
    for (int r = 0; r < 8; ++r) {
      lrow[r] = lrow[r] * alpha[r] + rs[r];
#pragma unroll
      for (int t = 0; t < 4; ++t) acc[t][r] *= alpha[r];
    }

    // --- P (C layout) -> per-wave LDS row-major [16][32] -> A layout ---
#pragma unroll
    for (int r = 0; r < 8; ++r) {
      int row = r + 8 * half;
      pw[row * 32 + m]      = (__bf16)s0[r];
      pw[row * 32 + 16 + m] = (__bf16)s1[r];
    }
    v16bf ap = ld16(pw + m * 32 + half * 16);

    // --- O += P * V ---
#pragma unroll
    for (int t = 0; t < 4; ++t) acc[t] = wmma_bf16(ap, bv[t], acc[t]);

    // stage next block into the other buffer, then sync
    if (more) {
      st8(&Kl[nxt][krow][kcol], kr);
      st8(&Vl[nxt][vrow][vcol], vr);
    }
    __syncthreads();
  }

  float inv[8];
#pragma unroll
  for (int r = 0; r < 8; ++r) inv[r] = __builtin_amdgcn_rcpf(lrow[r]);

  const int bb = bh >> 3, head = bh & 7;
  __bf16* orow = Ob + ((size_t)bb * NTOK + qtile * 16) * INNER + head * DH;
#pragma unroll
  for (int t = 0; t < 4; ++t)
#pragma unroll
    for (int r = 0; r < 8; ++r)
      orow[(size_t)(r + 8 * half) * INNER + t * 16 + m] = (__bf16)(acc[t][r] * inv[r]);
}

// ---------------- stage 3: output projection (out = O * Wo^T, f32 out) ----------------
// One wave -> 16x64 tile (4 c-tiles share each A operand), software-pipelined.
__global__ void __launch_bounds__(256) k_out(const __bf16* __restrict__ Ob,
                                             const __bf16* __restrict__ Wob,
                                             float* __restrict__ out) {
  const int lane = threadIdx.x & 31;
  const int wid  = blockIdx.x * 8 + (threadIdx.x >> 5);
  const int rowtile = wid >> 2;   // 512
  const int cg      = wid & 3;    // c-group of 64
  const int m = lane & 15, half = lane >> 4;

  const __bf16* orow = Ob + (size_t)(rowtile * 16 + m) * INNER;
  const __bf16* wr   = Wob + (size_t)(cg * 64 + m) * INNER + half * 16;
  v8f acc[4];
#pragma unroll
  for (int t = 0; t < 4; ++t) acc[t] = vzero();

  // pipeline prologue
  v16bf a = ld16(orow + half * 16);
  v16bf b[4];
#pragma unroll
  for (int t = 0; t < 4; ++t) b[t] = ld16(wr + (size_t)t * 16 * INNER);

#pragma unroll
  for (int e = 0; e < INNER; e += 32) {
    v16bf an, bn[4];
    if (e + 32 < INNER) {
      an = ld16(orow + e + 32 + half * 16);
#pragma unroll
      for (int t = 0; t < 4; ++t) bn[t] = ld16(wr + (size_t)t * 16 * INNER + e + 32);
    }
#pragma unroll
    for (int t = 0; t < 4; ++t) acc[t] = wmma_bf16(a, b[t], acc[t]);
    if (e + 32 < INNER) {
      a = an;
#pragma unroll
      for (int t = 0; t < 4; ++t) b[t] = bn[t];
    }
  }
#pragma unroll
  for (int t = 0; t < 4; ++t)
#pragma unroll
    for (int r = 0; r < 8; ++r)
      out[(size_t)(rowtile * 16 + r + 8 * half) * CCH + cg * 64 + t * 16 + m] = acc[t][r];
}

extern "C" void kernel_launch(void* const* d_in, const int* in_sizes, int n_in,
                              void* d_out, int out_size, void* d_ws, size_t ws_size,
                              hipStream_t stream) {
  (void)in_sizes; (void)n_in; (void)out_size; (void)ws_size;
  const float* x  = (const float*)d_in[0];
  const float* Wq = (const float*)d_in[1];
  const float* Wk = (const float*)d_in[2];
  const float* Wv = (const float*)d_in[3];
  const float* Wo = (const float*)d_in[4];
  float* out = (float*)d_out;

  char* ws = (char*)d_ws;
  size_t off = 0;
  __bf16* xb  = (__bf16*)(ws + off); off += (size_t)ROWS * CCH * 2;        // 4 MB
  __bf16* Wqb = (__bf16*)(ws + off); off += (size_t)INNER * CCH * 2;       // 256 KB
  __bf16* Wkb = (__bf16*)(ws + off); off += (size_t)INNER * CCH * 2;
  __bf16* Wvb = (__bf16*)(ws + off); off += (size_t)INNER * CCH * 2;
  __bf16* Wob = (__bf16*)(ws + off); off += (size_t)CCH * INNER * 2;
  __bf16* Qb  = (__bf16*)(ws + off); off += (size_t)BATCH * NH * NTOK * DH * 2;  // 8 MB
  __bf16* Kb  = (__bf16*)(ws + off); off += (size_t)BATCH * NH * NTOK * DH * 2;
  __bf16* Vtb = (__bf16*)(ws + off); off += (size_t)BATCH * NH * NTOK * DH * 2;
  __bf16* Ob  = (__bf16*)(ws + off); off += (size_t)ROWS * INNER * 2;      // 8 MB

  // stage 0: conversions
  int nx = ROWS * CCH;              // 2,097,152
  int nw = INNER * CCH;             // 131,072
  k_cvt_bf16<<<(nx + 255) / 256, 256, 0, stream>>>(x,  xb,  nx);
  k_cvt_bf16<<<(nw + 255) / 256, 256, 0, stream>>>(Wq, Wqb, nw);
  k_cvt_bf16<<<(nw + 255) / 256, 256, 0, stream>>>(Wk, Wkb, nw);
  k_cvt_bf16<<<(nw + 255) / 256, 256, 0, stream>>>(Wv, Wvb, nw);
  k_cvt_bf16<<<(nw + 255) / 256, 256, 0, stream>>>(Wo, Wob, nw);

  // stage 1: QKV projections — 3 * 512 * 8 wave-tiles (16x64 each), 8 waves/block
  k_qkv<<<(3 * 512 * 8) / 8, 256, 0, stream>>>(xb, Wqb, Wkb, Wvb, Qb, Kb, Vtb);

  // stage 2: attention — 16 (b,head) * 32 q-groups blocks of 8 waves
  k_attn<<<BATCH * NH * 32, 256, 0, stream>>>(Qb, Kb, Vtb, Ob);

  // stage 3: output projection — 512 * 4 wave-tiles (16x64 each)
  k_out<<<(512 * 4) / 8, 256, 0, stream>>>(Ob, Wob, out);
}
